// ResetLSTM_74680891343088
// MI455X (gfx1250) — matrix operand order repaired and checked by reading, not compiled, not run on hardware
//
#include <hip/hip_runtime.h>
#include <hip/hip_bf16.h>
#include <stdint.h>

// ---------------------------------------------------------------------------
// ResetLSTM forward for MI455X (gfx1250, wave32, WMMA).
//
// T=512, B=64, D=256, H=512.  gates = [x_t, h] @ W + b,  W:[768,2048].
// Strategy:
//   * bf16 WMMA (v_wmma_f32_16x16x32_bf16) with fp32 accumulation.
//   * Prep kernels (massively parallel): x -> bf16, W -> bf16 packed in
//     B-fragment layout (per-lane-contiguous 32B -> 2x b128 loads).
//   * Persistent kernel: 16 blocks x 256 threads = 128 waves.  Wave owns one
//     16x16 h-tile and computes its four gate tiles (i,g,f,o) so the LSTM
//     cell update happens fully in registers -> ONE grid barrier per step.
//   * h is ping-pong buffered in bf16 (written pre-masked with reset[t+1]);
//     c stays fp32 and is masked at read.  Whole working set is L2-resident
//     (192 MB L2), so per-step traffic (~3 MB packed W + tiny x/h) is cheap;
//     the critical path is the 512 dependent GEMMs + barriers.
// ---------------------------------------------------------------------------

#define T_STEPS 512
#define BATCH   64
#define DIN     256
#define HID     512
#define NBLK    16
#define NTHR    256

typedef __attribute__((ext_vector_type(16))) __bf16   bfx16;
typedef __attribute__((ext_vector_type(8)))  float    fx8;
typedef __attribute__((ext_vector_type(4)))  unsigned uintx4;

union BFrag { bfx16 v; unsigned u[8]; uintx4 q[2]; };

__device__ __forceinline__ unsigned short f2bf(float f) {
  // round-to-nearest-even fp32 -> bf16
  unsigned u = __float_as_uint(f);
  return (unsigned short)((u + 0x7FFFu + ((u >> 16) & 1u)) >> 16);
}

__device__ __forceinline__ float sigmoidf(float x) {
  return 1.0f / (1.0f + __expf(-x));
}

// ---------------------------------------------------------------------------
// Workspace init: barrier words, ping-pong h (bf16), c (fp32) all zeroed.
// Re-run every kernel_launch so graph replays are self-contained.
// ---------------------------------------------------------------------------
__global__ void k_init(unsigned* __restrict__ bar,
                       unsigned short* __restrict__ hb,
                       float* __restrict__ c) {
  int i = blockIdx.x * blockDim.x + threadIdx.x;
  if (i < 64) bar[i] = 0u;
  if (i < 2 * BATCH * HID) hb[i] = 0;          // both h buffers
  if (i < BATCH * HID) c[i] = 0.0f;
}

// x (fp32, T*B*D) -> bf16, same row-major layout.
__global__ void k_cvt_x(const float* __restrict__ x,
                        unsigned short* __restrict__ xb, int n) {
  int i = blockIdx.x * blockDim.x + threadIdx.x;
  if (i < n) xb[i] = f2bf(x[i]);
}

// Pack W [768,2048] fp32 into bf16 B-fragment layout:
//   tile ti = ntAbs*24 + kc  (ntAbs: 0..127 over the 2048 gate cols,
//                             kc: 0..23 over K=768 in chunks of 32)
//   dword d = lane*8 + v inside tile; lane 0-15 holds col n=ntAbs*16+lane,
//   K{0..7,16..23} pairwise; lanes 16-31 same col group, K{8..15,24..31}.
__global__ void k_pack_w(const float* __restrict__ W, unsigned* __restrict__ pw) {
  int idx = blockIdx.x * blockDim.x + threadIdx.x;
  if (idx >= (768 * 2048) / 2) return;
  int ti     = idx >> 8;
  int within = idx & 255;
  int lane   = within >> 3;
  int v      = within & 7;
  int ntAbs  = ti / 24;
  int kc     = ti % 24;
  int n = ntAbs * 16 + (lane & 15);
  int k = kc * 32 + ((lane >> 4) << 3) + ((v < 4) ? (v << 1) : (16 + ((v - 4) << 1)));
  unsigned lo = f2bf(W[(size_t)k * 2048 + n]);
  unsigned hi = f2bf(W[(size_t)(k + 1) * 2048 + n]);
  pw[idx] = lo | (hi << 16);
}

// ---------------------------------------------------------------------------
// Device-wide barrier: counter + generation, agent scope, fenced.
// ---------------------------------------------------------------------------
__device__ __forceinline__ void grid_sync(unsigned* bar) {
  __threadfence();          // publish this wave's stores device-wide
  __syncthreads();
  if (threadIdx.x == 0) {
    unsigned gen  = __hip_atomic_load(&bar[1], __ATOMIC_RELAXED, __HIP_MEMORY_SCOPE_AGENT);
    unsigned prev = __hip_atomic_fetch_add(&bar[0], 1u, __ATOMIC_ACQ_REL, __HIP_MEMORY_SCOPE_AGENT);
    if (prev == (unsigned)(NBLK - 1)) {
      __hip_atomic_store(&bar[0], 0u, __ATOMIC_RELAXED, __HIP_MEMORY_SCOPE_AGENT);
      __hip_atomic_fetch_add(&bar[1], 1u, __ATOMIC_RELEASE, __HIP_MEMORY_SCOPE_AGENT);
    } else {
      while (__hip_atomic_load(&bar[1], __ATOMIC_ACQUIRE, __HIP_MEMORY_SCOPE_AGENT) == gen) {
        __builtin_amdgcn_s_sleep(2);
      }
    }
  }
  __syncthreads();
  __threadfence();          // acquire: don't consume stale cached h
}

// ---------------------------------------------------------------------------
// Persistent recurrent kernel. 16 blocks x 256 thr (8 waves) = 128 waves.
// Wave gw: mT = gw&3 (batch tile), nT = gw>>2 (hidden tile, 0..31).
// ---------------------------------------------------------------------------
__global__ __launch_bounds__(NTHR)
void k_lstm(const int* __restrict__ reset,
            const float* __restrict__ bias,
            const unsigned short* __restrict__ xb,
            const unsigned* __restrict__ pw,
            unsigned short* __restrict__ hb,   // 2 x [64*512] bf16 ping-pong
            float* __restrict__ c,             // [64*512] fp32
            unsigned* __restrict__ bar,
            float* __restrict__ out)           // [T,B,H] fp32
{
  const int lane = threadIdx.x & 31;
  const int wave = threadIdx.x >> 5;
  const int gw   = blockIdx.x * 8 + wave;     // 0..127
  const int mT   = gw & 3;
  const int nT   = gw >> 2;
  const int m0   = mT << 4;
  const int n0   = nT << 4;
  const int lhi  = lane >> 4;                 // 0 | 1
  const int ln   = lane & 15;

  // bias per gate for this lane's output column (C/D layout: N = lane&15)
  const float bi  = bias[0 * HID + n0 + ln];
  const float bg  = bias[1 * HID + n0 + ln];
  const float bff = bias[2 * HID + n0 + ln] + 1.0f;   // forget bias +1
  const float bo  = bias[3 * HID + n0 + ln];

  for (int t = 0; t < T_STEPS; ++t) {
    fx8 acc[4] = {};   // i, g, f, o accumulators (16x16 fp32 tiles)
    const unsigned short* hcur = hb + ((t & 1) ? (size_t)(BATCH * HID) : 0);

    // ---- K = 0..255 : x_t part --------------------------------------------
    for (int kc = 0; kc < 8; ++kc) {
      BFrag a;
      // 16-bit A 16x32 layout: lanes0-15 K{0..7,16..23}, lanes16-31 K{8..15,24..31}
      const unsigned short* abase =
          xb + (((size_t)t * BATCH) + (m0 + ln)) * DIN + kc * 32 + lhi * 8;
#pragma unroll
      for (int p = 0; p < 8; ++p) {
        int k = (p < 4) ? (p << 1) : (16 + ((p - 4) << 1));
        a.u[p] = *reinterpret_cast<const unsigned*>(abase + k);
      }
#pragma unroll
      for (int g = 0; g < 4; ++g) {
        const uintx4* s = reinterpret_cast<const uintx4*>(
            pw + (((size_t)(g * 32 + nT) * 24 + kc) << 8) + (lane << 3));
        BFrag b; b.q[0] = s[0]; b.q[1] = s[1];
        acc[g] = __builtin_amdgcn_wmma_f32_16x16x32_bf16(
            false, a.v, false, b.v, (short)0, acc[g], false, false);
      }
    }
    // ---- K = 256..767 : recurrent h part ----------------------------------
    for (int kc = 8; kc < 24; ++kc) {
      BFrag a;
      const unsigned short* abase =
          hcur + (size_t)(m0 + ln) * HID + (kc - 8) * 32 + lhi * 8;
#pragma unroll
      for (int p = 0; p < 8; ++p) {
        int k = (p < 4) ? (p << 1) : (16 + ((p - 4) << 1));
        a.u[p] = *reinterpret_cast<const unsigned*>(abase + k);
      }
#pragma unroll
      for (int g = 0; g < 4; ++g) {
        const uintx4* s = reinterpret_cast<const uintx4*>(
            pw + (((size_t)(g * 32 + nT) * 24 + kc) << 8) + (lane << 3));
        BFrag b; b.q[0] = s[0]; b.q[1] = s[1];
        acc[g] = __builtin_amdgcn_wmma_f32_16x16x32_bf16(
            false, a.v, false, b.v, (short)0, acc[g], false, false);
      }
    }

    // ---- in-register LSTM cell update (this wave owns these 256 (b,j)) ----
    unsigned short* hnext = hb + ((t & 1) ? 0 : (size_t)(BATCH * HID));
    const int j = n0 + ln;
#pragma unroll
    for (int v = 0; v < 8; ++v) {
      const int brow = m0 + (lhi << 3) + v;    // C/D layout: M = v (+8 hi lanes)
      const float keep = 1.0f - (float)reset[t * BATCH + brow];
      const float iv = sigmoidf(acc[0][v] + bi);
      const float gv = tanhf  (acc[1][v] + bg);
      const float fv = sigmoidf(acc[2][v] + bff);
      const float ov = sigmoidf(acc[3][v] + bo);
      const float cold = c[brow * HID + j] * keep;     // reset applied to c here
      const float nc = fv * cold + iv * gv;
      c[brow * HID + j] = nc;
      const float nh = ov * tanhf(nc);
      out[((size_t)t * BATCH + brow) * HID + j] = nh;
      // pre-apply next step's reset mask to h, store bf16 for next GEMM
      const float keepn = (t + 1 < T_STEPS)
                              ? (1.0f - (float)reset[(t + 1) * BATCH + brow])
                              : 1.0f;
      hnext[brow * HID + j] = f2bf(nh * keepn);
    }

    grid_sync(bar);   // single barrier per step (h is ping-ponged)
  }
}

// ---------------------------------------------------------------------------
extern "C" void kernel_launch(void* const* d_in, const int* in_sizes, int n_in,
                              void* d_out, int out_size, void* d_ws, size_t ws_size,
                              hipStream_t stream) {
  (void)in_sizes; (void)n_in; (void)out_size; (void)ws_size;
  const float* x     = (const float*)d_in[0];   // [T,B,D]
  const int*   reset = (const int*)  d_in[1];   // [T,B]
  const float* W     = (const float*)d_in[2];   // [768,2048]
  const float* b     = (const float*)d_in[3];   // [2048]
  float* out = (float*)d_out;

  // workspace carve (total ~20.2 MB)
  char* w = (char*)d_ws;
  unsigned*       bar = (unsigned*)(w + 0);                       // 256 B
  unsigned short* hb  = (unsigned short*)(w + 256);               // 131072 B
  float*          c   = (float*)(w + 256 + 131072);               // 131072 B
  unsigned short* xb  = (unsigned short*)(w + 262400);            // 16777216 B
  unsigned*       pw  = (unsigned*)(w + 262400 + 16777216);       // 3145728 B

  k_init  <<<dim3(256),   dim3(256), 0, stream>>>(bar, hb, c);
  k_cvt_x <<<dim3(32768), dim3(256), 0, stream>>>(x, xb, T_STEPS * BATCH * DIN);
  k_pack_w<<<dim3(3072),  dim3(256), 0, stream>>>(W, pw);
  k_lstm  <<<dim3(NBLK),  dim3(NTHR), 0, stream>>>(reset, b, xb, pw, hb, c, bar, out);
}
